// BlockSparseFFN_87385404604843
// MI455X (gfx1250) — compile-verified, matrix operand order checked by reading, not run on hardware
//
#include <hip/hip_runtime.h>
#include <stdint.h>

// BlockSparseFFN forward for MI455X (gfx1250).
// bf16 WMMA + fp32 accumulation; GLOBAL_LOAD_ASYNC_TO_LDS for the gate/up
// A-tile, TDM (tensor_load_to_lds) for the down-projection A-tile.
//
// d_ws layout: [ mask: N*NB fp32 = 2 MB ][ x_bf16: N*D = 32 MB ][ hidden: N*I bf16 = 128 MB ]

#define N_TOK   8192
#define DMODEL  2048
#define IDIM    8192
#define RDIM    128
#define NBLK    64
#define TOPK    16

#define TM 128      // tokens per workgroup tile
#define TN 128      // features per workgroup tile (== BLOCK_SIZE)
#define TK 32       // bf16 WMMA K
#define LDSW 40     // padded LDS row stride in 16-bit elems (80 B, 16B-aligned)

typedef __attribute__((ext_vector_type(16))) __bf16       v16bf;
typedef __attribute__((ext_vector_type(8)))  float        v8f;
typedef __attribute__((ext_vector_type(4)))  unsigned int v4u;
typedef __attribute__((ext_vector_type(2)))  unsigned int v2u;
typedef __attribute__((ext_vector_type(8)))  unsigned int v8u;

static __device__ __forceinline__ unsigned short f2bf(float f) {
  unsigned int u = __float_as_uint(f);
  u += 0x7FFFu + ((u >> 16) & 1u);        // round-to-nearest-even
  return (unsigned short)(u >> 16);
}
static __device__ __forceinline__ unsigned int pack2bf(float a, float b) {
  return (unsigned int)f2bf(a) | ((unsigned int)f2bf(b) << 16);
}
static __device__ __forceinline__ v8f vzero8() {
  v8f z = {0.f, 0.f, 0.f, 0.f, 0.f, 0.f, 0.f, 0.f};
  return z;
}
static __device__ __forceinline__ float silu_f(float g) {
  return g * (1.f / (1.f + __expf(-g)));
}

// LDS byte offset of a shared-memory pointer: flat LDS aperture keeps the
// wave-relative LDS offset in addr[31:0] (ISA 10.2 aperture mapping).
static __device__ __forceinline__ unsigned int lds_off32(const void* p) {
  return (unsigned int)(uintptr_t)p;
}

// Async copy of one 16B chunk global -> LDS (GLOBAL_LOAD_ASYNC_TO_LDS_B128,
// tracked by ASYNCcnt). No VGPR round-trip for the data.
static __device__ __forceinline__ void async_cp16(unsigned int lds_off, const void* gptr) {
  asm volatile("global_load_async_to_lds_b128 %0, %1, off"
               :: "v"(lds_off), "v"((unsigned long long)(uintptr_t)gptr)
               : "memory");
}
static __device__ __forceinline__ void wait_async0() {
  asm volatile("s_wait_asynccnt 0x0" ::: "memory");
}

// TDM: move a 2-D tile (rows x 64B) global -> LDS with native LDS padding
// (pad 4 DWORDs after every 16 DWORDs => 80B row stride == LDSW). 2-D tensor
// => 2 descriptor groups (VADDR2/3 NULL). Tracked by TENSORcnt.
static __device__ __forceinline__ void tdm_load_tile_64B_rows(
    unsigned int lds_addr, unsigned long long gaddr,
    unsigned int rows, unsigned int row_stride_elems /* bf16 elems */) {
  v4u g0;
  g0[0] = 1u;                                   // count=1, user descriptor
  g0[1] = lds_addr;                             // lds_addr (bytes)
  g0[2] = (unsigned int)gaddr;                  // global_addr[31:0]
  g0[3] = ((unsigned int)(gaddr >> 32) & 0x01FFFFFFu) | (2u << 30);  // addr[56:32] | type=2
  v8u g1;
  g1[0] = (1u << 16)        // data_size = 2 bytes
        | (1u << 20)        // pad_enable
        | (3u << 22)        // pad_interval: 16 DWORDs (one 64B row)
        | (3u << 25);       // pad_amount:   4 DWORDs (16B)
  g1[1] = (IDIM & 0xFFFFu) << 16;                       // tensor_dim0[15:0] @ bit48
  g1[2] = (IDIM >> 16) | ((IDIM & 0xFFFFu) << 16);      // tensor_dim0[31:16] | tensor_dim1[15:0]
  g1[3] = (IDIM >> 16) | (32u << 16);                   // tensor_dim1[31:16] | tile_dim0 = 32 elems
  g1[4] = rows;                                         // tile_dim1 = rows, tile_dim2 = 0
  g1[5] = row_stride_elems;                             // tensor_dim0_stride[31:0]
  g1[6] = 0u;                                           // stride hi | dim1_stride lo
  g1[7] = 0u;
  asm volatile("tensor_load_to_lds %0, %1" :: "s"(g0), "s"(g1) : "memory");
}

// Assemble a 16x32 bf16 A/B fragment from a row-major LDS tile.
// Lanes 0-15 hold K=0..7 (v0..3) + K=16..23 (v4..7);
// lanes 16-31 hold K=8..15 + K=24..31. Caller passes p = rowbase + half*8.
union FragU { v4u q[2]; v16bf b; };
static __device__ __forceinline__ v16bf lds_frag(const unsigned short* p) {
  FragU r;
  r.q[0] = *(const v4u*)(p);        // 16 bytes
  r.q[1] = *(const v4u*)(p + 16);   // +32 bytes
  return r.b;
}

// ---------------------------------------------------------------------------
// One-shot fp32 -> bf16 cast of x (reused by all 64 feature-block columns).
// ---------------------------------------------------------------------------
__global__ __launch_bounds__(256) void xcast_kernel(
    const float* __restrict__ x, unsigned short* __restrict__ xb) {
  const size_t i = ((size_t)blockIdx.x * 256 + threadIdx.x) * 4;
  float4 v = *(const float4*)&x[i];
  *(v2u*)&xb[i] = (v2u){pack2bf(v.x, v.y), pack2bf(v.z, v.w)};
}

// ---------------------------------------------------------------------------
// Router: logits -> sigmoid probs + hard top-16 mask (STE forward).
// One workgroup (128 threads) per token.
// ---------------------------------------------------------------------------
__global__ __launch_bounds__(RDIM) void router_topk_kernel(
    const float* __restrict__ x, const float* __restrict__ w1,
    const float* __restrict__ w2, float* __restrict__ mask) {
  __shared__ float xs[DMODEL];
  __shared__ float h[RDIM];
  __shared__ float lg[NBLK];
  const int t = blockIdx.x;
  const int tid = threadIdx.x;

  for (int i = tid; i < DMODEL; i += RDIM) xs[i] = x[(size_t)t * DMODEL + i];
  __syncthreads();

  {  // h = silu(x @ w1^T), one row of w1 per thread
    const float4* x4 = (const float4*)xs;
    const float4* w4 = (const float4*)(w1 + (size_t)tid * DMODEL);
    float acc = 0.f;
    for (int d = 0; d < DMODEL / 4; ++d) {
      float4 a = x4[d], b = w4[d];
      acc += a.x * b.x + a.y * b.y + a.z * b.z + a.w * b.w;
    }
    h[tid] = silu_f(acc);
  }
  __syncthreads();

  if (tid < NBLK) {  // logits = h @ w2^T
    const float* w2r = w2 + (size_t)tid * RDIM;
    float a = 0.f;
    for (int j = 0; j < RDIM; ++j) a += w2r[j] * h[j];
    lg[tid] = a;
  }
  __syncthreads();

  if (tid < NBLK) {  // rank-based top-k, ties -> lower index (matches lax.top_k)
    float v = lg[tid];
    int rank = 0;
    for (int j = 0; j < NBLK; ++j) {
      float o = lg[j];
      rank += (o > v) || (o == v && j < tid);
    }
    float p  = 1.f / (1.f + __expf(-v));
    float hm = (rank < TOPK) ? 1.f : 0.f;
    mask[(size_t)t * NBLK + tid] = (hm - p) + p;  // forward of STE expression
  }
}

// ---------------------------------------------------------------------------
// Gate+Up fused GEMM: hidden[b, blk*128+c] = silu(x@G^T) * (x@U^T) * mask
// Tile 128 tokens x 128 features; 16 waves (4x4), each wave 32x32 per matrix.
// A-tile (x bf16) arrives via async global->LDS; weight tiles convert on stage.
// ---------------------------------------------------------------------------
__global__ __launch_bounds__(512) void gateup_wmma_kernel(
    const unsigned short* __restrict__ xb, const float* __restrict__ gw,
    const float* __restrict__ uw, const float* __restrict__ mask,
    unsigned short* __restrict__ hidden) {
  __shared__ __align__(16) unsigned short sx[TM * LDSW];
  __shared__ __align__(16) unsigned short sg[TN * LDSW];
  __shared__ __align__(16) unsigned short su[TN * LDSW];

  const int tid = threadIdx.x;
  const int wave = tid >> 5, lane = tid & 31;
  const int wm = wave >> 2, wn = wave & 3;
  const int half = lane >> 4, l16 = lane & 15;
  const int tok0 = blockIdx.x * TM;
  const int blk = blockIdx.y;  // intermediate block index (width 128)
  const int fb0 = blk * TN;

  // Per-thread async A-chunk mapping: one 16B chunk each (128 rows x 4 chunks).
  const int arow = tid >> 2, aq = tid & 3;
  const unsigned int a_lds = lds_off32(&sx[arow * LDSW + aq * 8]);

  v8f accG[2][2], accU[2][2];
  for (int i = 0; i < 2; ++i)
    for (int j = 0; j < 2; ++j) { accG[i][j] = vzero8(); accU[i][j] = vzero8(); }

  for (int kb = 0; kb < DMODEL; kb += TK) {
    // A tile: async bf16 copy, overlapped with B-tile conversion below.
    async_cp16(a_lds, &xb[(size_t)(tok0 + arow) * DMODEL + kb + aq * 8]);

    // B tiles: stage fp32 -> bf16 (2 float4 chunks per thread per matrix).
    for (int c = tid; c < TN * (TK / 4); c += 512) {
      const int row = c >> 3, f4 = c & 7;
      const size_t gofs = (size_t)kb + f4 * 4;
      float4 vg = *(const float4*)&gw[(size_t)(fb0 + row) * DMODEL + gofs];
      float4 vu = *(const float4*)&uw[(size_t)(fb0 + row) * DMODEL + gofs];
      *(v2u*)&sg[row * LDSW + f4 * 4] = (v2u){pack2bf(vg.x, vg.y), pack2bf(vg.z, vg.w)};
      *(v2u*)&su[row * LDSW + f4 * 4] = (v2u){pack2bf(vu.x, vu.y), pack2bf(vu.z, vu.w)};
      if (kb + TK < DMODEL) {  // prefetch next weight panel into cache
        __builtin_prefetch(&gw[(size_t)(fb0 + row) * DMODEL + gofs + TK], 0, 3);
        __builtin_prefetch(&uw[(size_t)(fb0 + row) * DMODEL + gofs + TK], 0, 3);
      }
    }
    wait_async0();
    __syncthreads();

    v16bf a0 = lds_frag(&sx[(wm * 32 +  0 + l16) * LDSW + half * 8]);
    v16bf a1 = lds_frag(&sx[(wm * 32 + 16 + l16) * LDSW + half * 8]);
    for (int nt = 0; nt < 2; ++nt) {
      v16bf bg = lds_frag(&sg[(wn * 32 + nt * 16 + l16) * LDSW + half * 8]);
      accG[0][nt] = __builtin_amdgcn_wmma_f32_16x16x32_bf16(false, a0, false, bg, (short)0, accG[0][nt], false, false);
      accG[1][nt] = __builtin_amdgcn_wmma_f32_16x16x32_bf16(false, a1, false, bg, (short)0, accG[1][nt], false, false);
      v16bf bu = lds_frag(&su[(wn * 32 + nt * 16 + l16) * LDSW + half * 8]);
      accU[0][nt] = __builtin_amdgcn_wmma_f32_16x16x32_bf16(false, a0, false, bu, (short)0, accU[0][nt], false, false);
      accU[1][nt] = __builtin_amdgcn_wmma_f32_16x16x32_bf16(false, a1, false, bu, (short)0, accU[1][nt], false, false);
    }
    __syncthreads();
  }

  // Epilogue: hidden = silu(gate) * up * mask[token][blk]   (C layout: M = r + 8*half)
  for (int mt = 0; mt < 2; ++mt) {
    const int rowbase = tok0 + wm * 32 + mt * 16 + half * 8;
    float m[8];
    for (int r = 0; r < 8; ++r) m[r] = mask[(size_t)(rowbase + r) * NBLK + blk];
    for (int nt = 0; nt < 2; ++nt) {
      const int col = fb0 + wn * 32 + nt * 16 + l16;
      for (int r = 0; r < 8; ++r) {
        float g = accG[mt][nt][r];
        float u = accU[mt][nt][r];
        hidden[(size_t)(rowbase + r) * IDIM + col] = f2bf(silu_f(g) * u * m[r]);
      }
    }
  }
}

// ---------------------------------------------------------------------------
// Down projection: out = hidden(bf16) @ down_w^T, fp32 out.
// A-tile (hidden bf16) via TDM tensor_load_to_lds (one descriptor per tile,
// LDS padding done by the data mover); weight tile converts on stage.
// ---------------------------------------------------------------------------
__global__ __launch_bounds__(512) void down_wmma_kernel(
    const unsigned short* __restrict__ hidden, const float* __restrict__ dw,
    float* __restrict__ out) {
  __shared__ __align__(16) unsigned short sh[TM * LDSW];
  __shared__ __align__(16) unsigned short sw[TN * LDSW];

  const int tid = threadIdx.x;
  const int wave = tid >> 5, lane = tid & 31;
  const int wm = wave >> 2, wn = wave & 3;
  const int half = lane >> 4, l16 = lane & 15;
  const int tok0 = blockIdx.x * TM;
  const int ob0 = blockIdx.y * TN;

  const unsigned int sh_lds = lds_off32(&sh[0]);
  const unsigned long long hid_base =
      (unsigned long long)(uintptr_t)hidden + (size_t)tok0 * IDIM * 2ull;

  v8f acc[2][2];
  for (int i = 0; i < 2; ++i)
    for (int j = 0; j < 2; ++j) acc[i][j] = vzero8();

  for (int kb = 0; kb < IDIM; kb += TK) {
    // A tile: single TDM descriptor moves 128 rows x 64B with 16B LDS padding
    // per row (matches the 80B padded LDS stride). Wave 0 issues; TDM is a
    // wave-level op (EXEC ignored), TENSORcnt tracks completion.
    if (wave == 0) {
      tdm_load_tile_64B_rows(sh_lds, hid_base + (size_t)kb * 2ull, TM, IDIM);
    }

    for (int c = tid; c < TN * (TK / 4); c += 512) {  // B tile: fp32 -> bf16
      const int row = c >> 3, f4 = c & 7;
      float4 vw = *(const float4*)&dw[(size_t)(ob0 + row) * IDIM + kb + f4 * 4];
      *(v2u*)&sw[row * LDSW + f4 * 4] = (v2u){pack2bf(vw.x, vw.y), pack2bf(vw.z, vw.w)};
      if (kb + TK < IDIM)
        __builtin_prefetch(&dw[(size_t)(ob0 + row) * IDIM + kb + TK + f4 * 4], 0, 3);
    }
    if (wave == 0) {
      __builtin_amdgcn_s_wait_tensorcnt(0);
    }
    __syncthreads();

    v16bf a0 = lds_frag(&sh[(wm * 32 +  0 + l16) * LDSW + half * 8]);
    v16bf a1 = lds_frag(&sh[(wm * 32 + 16 + l16) * LDSW + half * 8]);
    for (int nt = 0; nt < 2; ++nt) {
      v16bf b = lds_frag(&sw[(wn * 32 + nt * 16 + l16) * LDSW + half * 8]);
      acc[0][nt] = __builtin_amdgcn_wmma_f32_16x16x32_bf16(false, a0, false, b, (short)0, acc[0][nt], false, false);
      acc[1][nt] = __builtin_amdgcn_wmma_f32_16x16x32_bf16(false, a1, false, b, (short)0, acc[1][nt], false, false);
    }
    __syncthreads();
  }

  for (int mt = 0; mt < 2; ++mt) {
    const int rowbase = tok0 + wm * 32 + mt * 16 + half * 8;
    for (int nt = 0; nt < 2; ++nt) {
      const int col = ob0 + wn * 32 + nt * 16 + l16;
      for (int r = 0; r < 8; ++r)
        out[(size_t)(rowbase + r) * DMODEL + col] = acc[mt][nt][r];
    }
  }
}

// ---------------------------------------------------------------------------
extern "C" void kernel_launch(void* const* d_in, const int* in_sizes, int n_in,
                              void* d_out, int out_size, void* d_ws, size_t ws_size,
                              hipStream_t stream) {
  (void)in_sizes; (void)n_in; (void)out_size; (void)ws_size;
  const float* x   = (const float*)d_in[0];
  const float* gw  = (const float*)d_in[1];
  const float* uw  = (const float*)d_in[2];
  const float* dw  = (const float*)d_in[3];
  const float* rw1 = (const float*)d_in[4];
  const float* rw2 = (const float*)d_in[5];
  float* out = (float*)d_out;

  char* ws = (char*)d_ws;
  float* mask = (float*)ws;                                    // N*NB fp32 (2 MB)
  unsigned short* xbf = (unsigned short*)(ws + (size_t)N_TOK * NBLK * sizeof(float));
  unsigned short* hidden = (unsigned short*)((char*)xbf + (size_t)N_TOK * DMODEL * 2);

  xcast_kernel<<<dim3((N_TOK * DMODEL / 4) / 256), dim3(256), 0, stream>>>(x, xbf);
  router_topk_kernel<<<dim3(N_TOK), dim3(RDIM), 0, stream>>>(x, rw1, rw2, mask);
  gateup_wmma_kernel<<<dim3(N_TOK / TM, IDIM / TN), dim3(512), 0, stream>>>(
      xbf, gw, uw, mask, hidden);
  down_wmma_kernel<<<dim3(N_TOK / TM, DMODEL / TN), dim3(512), 0, stream>>>(
      hidden, dw, out);
}
